// HeteroGNN_88046829568691
// MI455X (gfx1250) — compile-verified, matrix operand order checked by reading, not run on hardware
//
#include <hip/hip_runtime.h>
#include <hip/hip_bf16.h>

#define H 128                    // hidden dim
#define KTOT 256                 // 2*H concat K
#define NGRAPH 32
#define STRIPS 2                 // row-strips per wave per block (amortize W staging)

typedef __attribute__((ext_vector_type(16))) _Float16 v16h;
typedef __attribute__((ext_vector_type(8)))  float    v8f;

// ---------------------------------------------------------------------------
// zero-fill
// ---------------------------------------------------------------------------
__global__ void fillz_kernel(float* __restrict__ p, long n) {
    long i = (long)blockIdx.x * blockDim.x + threadIdx.x;
    long stride = (long)gridDim.x * blockDim.x;
    for (; i < n; i += stride) p[i] = 0.0f;
}

// ---------------------------------------------------------------------------
// scatter-add: one wave per edge, lane handles 4 features (float4).
// agg[dst] += feat[src];  cnt[dst] += 1
// ---------------------------------------------------------------------------
__global__ __launch_bounds__(256) void scatter_add_kernel(
    const float* __restrict__ feat, const int* __restrict__ sidx,
    const int* __restrict__ didx, float* __restrict__ agg,
    float* __restrict__ cnt, int nEdges)
{
    int gw   = (int)((blockIdx.x * 256 + threadIdx.x) >> 5);
    int lane = threadIdx.x & 31;
    int nw   = (int)((gridDim.x * 256) >> 5);
    for (int e = gw; e < nEdges; e += nw) {
        int s = sidx[e];
        int d = didx[e];
        const float4 v = *(const float4*)(feat + (size_t)s * H + lane * 4);
        float* dp = agg + (size_t)d * H + lane * 4;
        atomicAdd(dp + 0, v.x);
        atomicAdd(dp + 1, v.y);
        atomicAdd(dp + 2, v.z);
        atomicAdd(dp + 3, v.w);
        if (lane == 0) atomicAdd(cnt + d, 1.0f);
    }
}

// ---------------------------------------------------------------------------
// Fused: out[r,:] = ReLU( concat(agg[r]/max(cnt,1), xold[r]) @ W + b )
// W: [256,128] row-major f32, staged once per block into LDS as f16,
// pre-swizzled into WMMA B-fragment layout:
//   Wsh[((kt*8+nt)*32 + lane)*16 + i] = W[kt*32 + K(i,lane)][nt*16 + (lane&15)]
// K(i): j=i/2, hk=8*(lane>>4), K = (j<4 ? hk+2j : 16+hk+2(j-4)) + (i&1)
// 256-thread block = 8 waves; each wave computes STRIPS strips of 16 rows,
// reusing the staged weights. B fragments are double-buffered so ds_loads
// overlap the WMMA pipeline.
// ---------------------------------------------------------------------------
__global__ __launch_bounds__(256) void gemm_bias_relu_kernel(
    const float* __restrict__ agg, const float* __restrict__ cnt,
    const float* __restrict__ xold, const float* __restrict__ W,
    const float* __restrict__ bias, float* __restrict__ out, int R)
{
    __shared__ __attribute__((aligned(32))) _Float16 Wsh[8 * 8 * 32 * 16]; // 64 KB

    const int tid = threadIdx.x;

    // ---- stage + swizzle weights into LDS (f32 -> f16) ----
    for (int idx = tid; idx < 8 * 8 * 32 * 16; idx += 256) {
        int i    = idx & 15;
        int lane = (idx >> 4) & 31;
        int nt   = (idx >> 9) & 7;
        int kt   = idx >> 12;
        int hk   = (lane >> 4) * 8;
        int j    = i >> 1;
        int kk   = ((j < 4) ? (hk + 2 * j) : (16 + hk + 2 * (j - 4))) + (i & 1);
        Wsh[idx] = (_Float16)W[(kt * 32 + kk) * H + nt * 16 + (lane & 15)];
    }
    __syncthreads();

    const int wave = tid >> 5;
    const int lane = tid & 31;
    const int hk   = (lane >> 4) * 8;
    const int nlo  = lane & 15;

    #pragma unroll
    for (int strip = 0; strip < STRIPS; ++strip) {
        const int rowBase = (blockIdx.x * STRIPS + strip) * 128 + wave * 16;
        if (rowBase >= R) continue;
        const int arow  = rowBase + nlo;
        const int arowc = (arow < R) ? arow : (R - 1);

        const float scale = 1.0f / fmaxf(cnt[arowc], 1.0f);

        v8f acc[8] = {};

        #pragma unroll
        for (int kt = 0; kt < 8; ++kt) {
            // ---- A fragment: 2x 8 consecutive f32 per lane, scaled, cvt f16
            const float* src;
            float s;
            if (kt < 4) { src = agg  + (size_t)arowc * H + kt * 32;       s = scale; }
            else        { src = xold + (size_t)arowc * H + (kt - 4) * 32; s = 1.0f;  }

            v16h a;
            #pragma unroll
            for (int j = 0; j < 8; ++j) {
                a[j]     = (_Float16)(src[hk + j] * s);
                a[8 + j] = (_Float16)(src[16 + hk + j] * s);
            }

            // ---- 8 n-tiles, B double-buffered from LDS
            v16h b = *(const v16h*)&Wsh[((kt * 8 + 0) * 32 + lane) * 16];
            #pragma unroll
            for (int nt = 0; nt < 8; ++nt) {
                v16h bnext;
                if (nt < 7)
                    bnext = *(const v16h*)&Wsh[((kt * 8 + nt + 1) * 32 + lane) * 16];
                acc[nt] = __builtin_amdgcn_wmma_f32_16x16x32_f16(
                    false, a, false, b, (short)0, acc[nt], false, false);
                if (nt < 7) b = bnext;
            }
        }

        // ---- epilogue: bias + ReLU + store (D: row = v + 8*(lane>>4), col = lane&15)
        #pragma unroll
        for (int nt = 0; nt < 8; ++nt) {
            int n = nt * 16 + nlo;
            float bn = bias[n];
            #pragma unroll
            for (int v = 0; v < 8; ++v) {
                int r = rowBase + v + 8 * (lane >> 4);
                if (r < R) out[(size_t)r * H + n] = fmaxf(acc[nt][v] + bn, 0.0f);
            }
        }
    }
}

// ---------------------------------------------------------------------------
// graph mean-pool: LDS partial sums per block, one atomic flush per block
// ---------------------------------------------------------------------------
__global__ __launch_bounds__(256) void pool_kernel(
    const float* __restrict__ xl, const int* __restrict__ batch,
    float* __restrict__ accum, float* __restrict__ cnt, int nLink)
{
    __shared__ float sacc[NGRAPH * H];   // 16 KB
    __shared__ float scnt[NGRAPH];
    for (int i = threadIdx.x; i < NGRAPH * H; i += 256) sacc[i] = 0.0f;
    if (threadIdx.x < NGRAPH) scnt[threadIdx.x] = 0.0f;
    __syncthreads();

    int wave = threadIdx.x >> 5, lane = threadIdx.x & 31;
    int gw = blockIdx.x * 8 + wave;
    int nw = gridDim.x * 8;
    for (int l = gw; l < nLink; l += nw) {
        int g = batch[l];
        const float4 v = *(const float4*)(xl + (size_t)l * H + lane * 4);
        float* dp = sacc + g * H + lane * 4;
        atomicAdd(dp + 0, v.x);
        atomicAdd(dp + 1, v.y);
        atomicAdd(dp + 2, v.z);
        atomicAdd(dp + 3, v.w);
        if (lane == 0) atomicAdd(&scnt[g], 1.0f);
    }
    __syncthreads();

    for (int i = threadIdx.x; i < NGRAPH * H; i += 256) atomicAdd(&accum[i], sacc[i]);
    if (threadIdx.x < NGRAPH) atomicAdd(&cnt[threadIdx.x], scnt[threadIdx.x]);
}

__global__ void pool_final_kernel(const float* __restrict__ accum,
                                  const float* __restrict__ cnt,
                                  float* __restrict__ out)
{
    int i = blockIdx.x * blockDim.x + threadIdx.x;
    if (i < NGRAPH * H) out[i] = accum[i] / fmaxf(cnt[i >> 7], 1.0f);
}

// ---------------------------------------------------------------------------
// host launcher
// ---------------------------------------------------------------------------
extern "C" void kernel_launch(void* const* d_in, const int* in_sizes, int n_in,
                              void* d_out, int out_size, void* d_ws, size_t ws_size,
                              hipStream_t stream)
{
    const float* x_node = (const float*)d_in[0];
    const float* x_link = (const float*)d_in[1];
    const int*   nl_src = (const int*)d_in[2];
    const int*   nl_dst = (const int*)d_in[3];
    const int*   ln_src = (const int*)d_in[4];
    const int*   ln_dst = (const int*)d_in[5];
    const int*   batch  = (const int*)d_in[6];
    const float* W_nl   = (const float*)d_in[7];
    const float* b_nl   = (const float*)d_in[8];
    const float* W_ln   = (const float*)d_in[9];
    const float* b_ln   = (const float*)d_in[10];
    float* out = (float*)d_out;

    const int nNode   = in_sizes[0] / H;
    const int nLink   = in_sizes[1] / H;
    const int nEdge   = in_sizes[2];
    const int nLayers = in_sizes[8] / H;

    // ---- workspace carve-out (256B aligned slices) ----
    char* wsp = (char*)d_ws;
    auto carve = [&](size_t bytes) -> void* {
        void* p = wsp;
        wsp += (bytes + 255) & ~(size_t)255;
        return p;
    };
    float* xn[2]  = { (float*)carve((size_t)nNode * H * 4),
                      (float*)carve((size_t)nNode * H * 4) };
    float* xl[2]  = { (float*)carve((size_t)nLink * H * 4),
                      (float*)carve((size_t)nLink * H * 4) };
    float* aggL   = (float*)carve((size_t)nLink * H * 4);
    float* aggN   = (float*)carve((size_t)nNode * H * 4);
    float* cntL   = (float*)carve((size_t)nLink * 4);
    float* cntN   = (float*)carve((size_t)nNode * 4);
    float* paccum = (float*)carve((size_t)NGRAPH * H * 4);
    float* pcnt   = (float*)carve((size_t)NGRAPH * 4);

    hipMemcpyAsync(xn[0], x_node, (size_t)nNode * H * 4, hipMemcpyDeviceToDevice, stream);
    hipMemcpyAsync(xl[0], x_link, (size_t)nLink * H * 4, hipMemcpyDeviceToDevice, stream);

    const int rowsPerBlock = 128 * STRIPS;
    int cur = 0;
    for (int i = 0; i < nLayers; ++i) {
        fillz_kernel<<<2048, 256, 0, stream>>>(aggL, (long)nLink * H);
        fillz_kernel<<<256,  256, 0, stream>>>(cntL, (long)nLink);
        fillz_kernel<<<2048, 256, 0, stream>>>(aggN, (long)nNode * H);
        fillz_kernel<<<256,  256, 0, stream>>>(cntN, (long)nNode);

        // ('node','nl','link'): gather node feats, mean-agg at links
        scatter_add_kernel<<<4096, 256, 0, stream>>>(xn[cur], nl_src, nl_dst,
                                                     aggL, cntL, nEdge);
        // ('link','ln','node'): gather (old) link feats, mean-agg at nodes
        scatter_add_kernel<<<4096, 256, 0, stream>>>(xl[cur], ln_src, ln_dst,
                                                     aggN, cntN, nEdge);

        gemm_bias_relu_kernel<<<(nLink + rowsPerBlock - 1) / rowsPerBlock, 256, 0, stream>>>(
            aggL, cntL, xl[cur], W_nl + (size_t)i * KTOT * H, b_nl + (size_t)i * H,
            xl[1 - cur], nLink);
        gemm_bias_relu_kernel<<<(nNode + rowsPerBlock - 1) / rowsPerBlock, 256, 0, stream>>>(
            aggN, cntN, xn[cur], W_ln + (size_t)i * KTOT * H, b_ln + (size_t)i * H,
            xn[1 - cur], nNode);

        cur ^= 1;
    }

    fillz_kernel<<<16, 256, 0, stream>>>(paccum, (long)NGRAPH * H);
    fillz_kernel<<<1,  256, 0, stream>>>(pcnt, (long)NGRAPH);
    pool_kernel<<<512, 256, 0, stream>>>(xl[cur], batch, paccum, pcnt, nLink);
    pool_final_kernel<<<(NGRAPH * H + 255) / 256, 256, 0, stream>>>(paccum, pcnt, out);
}
